// energy_block_68427418960432
// MI455X (gfx1250) — compile-verified, matrix operand order checked by reading, not run on hardware
//
#include <hip/hip_runtime.h>
#include <hip/hip_bf16.h>
#include <math.h>

typedef __attribute__((ext_vector_type(16))) _Float16 v16h;
typedef __attribute__((ext_vector_type(4)))  _Float16 v4h;
typedef __attribute__((ext_vector_type(8)))  float    v8f;

#define N_CNT   24576
#define E_CNT   565248
#define D_H     128
#define B_CNT   1024
#define NN_CNT  24
#define K1_PAD  320     // 313 padded to multiple of 32
#define LN_EPS  1e-5f

// ---------------------------------------------------------------------------
// helpers (fast rcp avoids v_div_scale fixup chains)
// ---------------------------------------------------------------------------
__device__ __forceinline__ float silu_f(float v) {
    return v * __builtin_amdgcn_rcpf(1.0f + __expf(-v));
}
__device__ __forceinline__ float sigmoid_f(float v) {
    return __builtin_amdgcn_rcpf(1.0f + __expf(-v));
}

// A fragment (16x32 f16, MxK) from LDS row-major [16][ldk] at column k0.
// lane<16: row=lane, elems 0..7 -> K=k0+0..7, elems 8..15 -> K=k0+16..23
// lane>=16: row=lane-16, K offsets +8. Two aligned ds_load_b128 per fragment.
__device__ __forceinline__ v16h load_frag_a(const _Float16* lds, int ldk, int k0, int lane) {
    int r  = lane & 15;
    int kb = k0 + ((lane >> 4) << 3);
    const _Float16* p = lds + r * ldk + kb;
    v16h a;
#pragma unroll
    for (int e = 0; e < 8; ++e) a[e] = p[e];
#pragma unroll
    for (int e = 0; e < 8; ++e) a[8 + e] = p[16 + e];
    return a;
}

// B fragment from fragment-swizzled weights: [kt][nt][lane][16] contiguous f16.
// One aligned 32-byte per-lane read (2x global_load_b128), no packing VALU.
__device__ __forceinline__ v16h load_frag_b_sw(const _Float16* w, int ntiles,
                                               int kt, int nt, int lane) {
    return ((const v16h*)w)[(kt * ntiles + nt) * 32 + lane];
}

// ---------------------------------------------------------------------------
// weight conversion: f32 [Ksrc][Nw] -> f16 fragment-swizzled [Kpad/32][Nw/16][32][16]
// ---------------------------------------------------------------------------
__global__ void cvt_swz_kernel(const float* __restrict__ src, _Float16* __restrict__ dst,
                               int Ksrc, int Nw, int Kpad) {
    int i = blockIdx.x * blockDim.x + threadIdx.x;
    int tot = Kpad * Nw;
    if (i >= tot) return;
    int ntiles = Nw >> 4;
    int e    = i & 15;
    int lane = (i >> 4) & 31;
    int rest = i >> 9;
    int nt   = rest % ntiles;
    int kt   = rest / ntiles;
    int hi   = lane >> 4;
    int k = kt * 32 + hi * 8 + ((e < 8) ? e : e + 8);
    int n = nt * 16 + (lane & 15);
    dst[i] = (k < Ksrc) ? (_Float16)src[k * Nw + n] : (_Float16)0.0f;
}

__global__ void zero_kernel(float* __restrict__ p, int n) {
    int i = blockIdx.x * blockDim.x + threadIdx.x;
    if (i < n) p[i] = 0.0f;
}

// ---------------------------------------------------------------------------
// edge message kernel: 2 waves/block, each wave owns a 16-edge tile.
// ---------------------------------------------------------------------------
__launch_bounds__(64)
__global__ void edge_msg_kernel(const float* __restrict__ x, const float* __restrict__ h,
                                const int* __restrict__ ei,
                                const float* __restrict__ bond, const float* __restrict__ dis,
                                const _Float16* __restrict__ W1h, const float* __restrict__ b1,
                                const float* __restrict__ ln1w, const float* __restrict__ ln1b,
                                const _Float16* __restrict__ W2h, const float* __restrict__ b2,
                                const float* __restrict__ ln2w, const float* __restrict__ ln2b,
                                float* __restrict__ agg) {
    __shared__ _Float16 sA[2][16 * K1_PAD];   // 20 KB
    __shared__ float    sC[2][16 * 256];      // 32 KB

    const int wave = threadIdx.x >> 5;
    const int lane = threadIdx.x & 31;
    const int tile = blockIdx.x * 2 + wave;
    const int e0   = tile * 16;

    const int* rowp = ei;            // edge_index[0]
    const int* colp = ei + E_CNT;    // edge_index[1]

    _Float16* A = sA[wave];
    float*    C = sC[wave];

    // ---- frame features (lanes 0..15, one edge each) + pad ----
    if (lane < 16) {
        int e = e0 + lane;
        int r = rowp[e], c = colp[e];
        float xr0 = x[3 * r], xr1 = x[3 * r + 1], xr2 = x[3 * r + 2];
        float xc0 = x[3 * c], xc1 = x[3 * c + 1], xc2 = x[3 * c + 2];
        float dx0 = xr0 - xc0, dx1 = xr1 - xc1, dx2 = xr2 - xc2;
        float cr0 = xr1 * xc2 - xr2 * xc1;
        float cr1 = xr2 * xc0 - xr0 * xc2;
        float cr2 = xr0 * xc1 - xr1 * xc0;
        float dn = __builtin_amdgcn_rcpf(sqrtf(dx0 * dx0 + dx1 * dx1 + dx2 * dx2 + 3e-30f));
        dx0 *= dn; dx1 *= dn; dx2 *= dn;
        float cn = __builtin_amdgcn_rcpf(sqrtf(cr0 * cr0 + cr1 * cr1 + cr2 * cr2 + 3e-30f));
        cr0 *= cn; cr1 *= cn; cr2 *= cn;
        float v0 = dx1 * cr2 - dx2 * cr1;
        float v1 = dx2 * cr0 - dx0 * cr2;
        float v2 = dx0 * cr1 - dx1 * cr0;
        float fi0 = xr0 * dx0 + xr1 * dx1 + xr2 * dx2;
        float fi1 = xr0 * cr0 + xr1 * cr1 + xr2 * cr2;
        float fi2 = xr0 * v0  + xr1 * v1  + xr2 * v2;
        float fj0 = xc0 * dx0 + xc1 * dx1 + xc2 * dx2;
        float fj1 = xc0 * cr0 + xc1 * cr1 + xc2 * cr2;
        float fj2 = xc0 * v0  + xc1 * v1  + xc2 * v2;
        float d0 = fi0 - fj0, d1 = fi1 - fj1, d2 = fi2 - fj2;
        float radial = d0 * d0 + d1 * d1 + d2 * d2;
        float ni = sqrtf(fi0 * fi0 + fi1 * fi1 + fi2 * fi2);
        float nj = sqrtf(fj0 * fj0 + fj1 * fj1 + fj2 * fj2);
        float cosv = (fi0 * fj0 + fi1 * fj1 + fi2 * fj2) *
                     __builtin_amdgcn_rcpf(fmaxf(ni, 1e-8f) * fmaxf(nj, 1e-8f));
        cosv = fminf(fmaxf(cosv, -1.0f), 1.0f);
        float sinv = sqrtf(1.0f - cosv * cosv + 1e-30f);
        _Float16* row_ff = A + lane * K1_PAD + 256;
        row_ff[0] = (_Float16)fi0; row_ff[1] = (_Float16)fi1; row_ff[2] = (_Float16)fi2;
        row_ff[3] = (_Float16)fj0; row_ff[4] = (_Float16)fj1; row_ff[5] = (_Float16)fj2;
        row_ff[6] = (_Float16)radial; row_ff[7] = (_Float16)cosv; row_ff[8] = (_Float16)sinv;
#pragma unroll
        for (int p = 313; p < K1_PAD; ++p) A[lane * K1_PAD + p] = (_Float16)0.0f;
    }

    // ---- gather h[row], h[col] into A: float4 loads + packed 8B LDS stores ----
    for (int idx = lane; idx < 16 * 32; idx += 32) {
        int er = idx >> 5, dq = (idx & 31) << 2;
        int e  = e0 + er;
        int r  = rowp[e], c = colp[e];
        float4 hr = *(const float4*)(h + r * 128 + dq);
        float4 hc = *(const float4*)(h + c * 128 + dq);
        v4h pr, pc;
        pr[0] = (_Float16)hr.x; pr[1] = (_Float16)hr.y; pr[2] = (_Float16)hr.z; pr[3] = (_Float16)hr.w;
        pc[0] = (_Float16)hc.x; pc[1] = (_Float16)hc.y; pc[2] = (_Float16)hc.z; pc[3] = (_Float16)hc.w;
        *(v4h*)(A + er * K1_PAD + dq)       = pr;
        *(v4h*)(A + er * K1_PAD + 128 + dq) = pc;
    }
    // ---- bond(16) + dis(32) (odd base offset 265 -> scalar stores) ----
    for (int idx = lane; idx < 16 * 48; idx += 32) {
        int er = idx / 48, d = idx % 48;
        int e  = e0 + er;
        float v = (d < 16) ? bond[e * 16 + d] : dis[e * 32 + (d - 16)];
        A[er * K1_PAD + 265 + d] = (_Float16)v;
    }
    __syncthreads();

    const int r0 = (lane >> 4) << 3;
    const int nc = lane & 15;

    // ---- layer 1: [16,320] @ [320,256], dual N-tiles for ILP ----
    for (int nt = 0; nt < 16; nt += 2) {
        v8f acc0 = {}, acc1 = {};
#pragma unroll
        for (int kt = 0; kt < 10; ++kt) {
            v16h a  = load_frag_a(A, K1_PAD, kt * 32, lane);
            v16h b0 = load_frag_b_sw(W1h, 16, kt, nt, lane);
            v16h b1 = load_frag_b_sw(W1h, 16, kt, nt + 1, lane);
            acc0 = __builtin_amdgcn_wmma_f32_16x16x32_f16(false, a, false, b0,
                                                          (short)0, acc0, false, false);
            acc1 = __builtin_amdgcn_wmma_f32_16x16x32_f16(false, a, false, b1,
                                                          (short)0, acc1, false, false);
        }
        int c0 = nt * 16 + nc, c1 = c0 + 16;
        float bb0 = b1[c0], bb1 = b1[c1];
#pragma unroll
        for (int e = 0; e < 8; ++e) {
            C[(r0 + e) * 256 + c0] = silu_f(acc0[e] + bb0);
            C[(r0 + e) * 256 + c1] = silu_f(acc1[e] + bb1);
        }
    }
    __syncthreads();

    // ---- LN over 256 (lanes 0..15, float4 scans), f16 A2 into reused A ----
    if (lane < 16) {
        const float4* cr4 = (const float4*)(C + lane * 256);
        float mu = 0.0f;
        for (int q = 0; q < 64; ++q) {
            float4 t = cr4[q]; mu += t.x + t.y + t.z + t.w;
        }
        mu *= (1.0f / 256.0f);
        float var = 0.0f;
        for (int q = 0; q < 64; ++q) {
            float4 t = cr4[q];
            float a0 = t.x - mu, a1 = t.y - mu, a2 = t.z - mu, a3 = t.w - mu;
            var += a0 * a0 + a1 * a1 + a2 * a2 + a3 * a3;
        }
        var *= (1.0f / 256.0f);
        float inv = rsqrtf(var + LN_EPS);
        _Float16* ar = A + lane * 256;
        const float4* w4 = (const float4*)ln1w;
        const float4* bb4 = (const float4*)ln1b;
        for (int q = 0; q < 64; ++q) {
            float4 t = cr4[q], w = w4[q], b = bb4[q];
            v4h o;
            o[0] = (_Float16)((t.x - mu) * inv * w.x + b.x);
            o[1] = (_Float16)((t.y - mu) * inv * w.y + b.y);
            o[2] = (_Float16)((t.z - mu) * inv * w.z + b.z);
            o[3] = (_Float16)((t.w - mu) * inv * w.w + b.w);
            *(v4h*)(ar + q * 4) = o;
        }
    }
    __syncthreads();

    // ---- layer 2: [16,256] @ [256,128] ----
    for (int nt = 0; nt < 8; nt += 2) {
        v8f acc0 = {}, acc1 = {};
#pragma unroll
        for (int kt = 0; kt < 8; ++kt) {
            v16h a  = load_frag_a(A, 256, kt * 32, lane);
            v16h b0 = load_frag_b_sw(W2h, 8, kt, nt, lane);
            v16h b1 = load_frag_b_sw(W2h, 8, kt, nt + 1, lane);
            acc0 = __builtin_amdgcn_wmma_f32_16x16x32_f16(false, a, false, b0,
                                                          (short)0, acc0, false, false);
            acc1 = __builtin_amdgcn_wmma_f32_16x16x32_f16(false, a, false, b1,
                                                          (short)0, acc1, false, false);
        }
        int c0 = nt * 16 + nc, c1 = c0 + 16;
        float bb0 = b2[c0], bb1 = b2[c1];
#pragma unroll
        for (int e = 0; e < 8; ++e) {
            C[(r0 + e) * 128 + c0] = silu_f(acc0[e] + bb0);
            C[(r0 + e) * 128 + c1] = silu_f(acc1[e] + bb1);
        }
    }
    __syncthreads();

    // ---- LN over 128 in-place ----
    if (lane < 16) {
        float4* cr4 = (float4*)(C + lane * 128);
        float mu = 0.0f;
        for (int q = 0; q < 32; ++q) {
            float4 t = cr4[q]; mu += t.x + t.y + t.z + t.w;
        }
        mu *= (1.0f / 128.0f);
        float var = 0.0f;
        for (int q = 0; q < 32; ++q) {
            float4 t = cr4[q];
            float a0 = t.x - mu, a1 = t.y - mu, a2 = t.z - mu, a3 = t.w - mu;
            var += a0 * a0 + a1 * a1 + a2 * a2 + a3 * a3;
        }
        var *= (1.0f / 128.0f);
        float inv = rsqrtf(var + LN_EPS);
        const float4* w4 = (const float4*)ln2w;
        const float4* b4 = (const float4*)ln2b;
        for (int q = 0; q < 32; ++q) {
            float4 t = cr4[q], w = w4[q], b = b4[q];
            t.x = (t.x - mu) * inv * w.x + b.x;
            t.y = (t.y - mu) * inv * w.y + b.y;
            t.z = (t.z - mu) * inv * w.z + b.z;
            t.w = (t.w - mu) * inv * w.w + b.w;
            cr4[q] = t;
        }
    }
    __syncthreads();

    // ---- scatter-add to agg (f32 global atomics, L2-resident) ----
    for (int idx = lane; idx < 16 * 128; idx += 32) {
        int er = idx >> 7, d = idx & 127;
        int r  = rowp[e0 + er];
        __hip_atomic_fetch_add(&agg[r * 128 + d], C[er * 128 + d],
                               __ATOMIC_RELAXED, __HIP_MEMORY_SCOPE_AGENT);
    }
}

// ---------------------------------------------------------------------------
// node kernel: 2 waves/block, each wave owns a 16-node tile (contiguous ids).
// ---------------------------------------------------------------------------
__launch_bounds__(64)
__global__ void node_kernel(const float* __restrict__ h, const float* __restrict__ agg,
                            const _Float16* __restrict__ Wnh, const float* __restrict__ bn,
                            const float* __restrict__ Wa, const float* __restrict__ ba,
                            const _Float16* __restrict__ Wp1h, const float* __restrict__ bp1,
                            const float* __restrict__ lnpw, const float* __restrict__ lnpb,
                            const float* __restrict__ Wp2,
                            float* __restrict__ neh_out, float* __restrict__ node_energy) {
    __shared__ _Float16 sA[2][16 * 256];   // 16 KB
    __shared__ float    sC[2][16 * 128];   // 16 KB

    const int wave = threadIdx.x >> 5;
    const int lane = threadIdx.x & 31;
    const int tile = blockIdx.x * 2 + wave;
    const int n0   = tile * 16;

    _Float16* A = sA[wave];
    float*    C = sC[wave];

    // ---- pack [h | agg] f16 A tile (vectorized) ----
    for (int idx = lane; idx < 16 * 32; idx += 32) {
        int r = idx >> 5, dq = (idx & 31) << 2;
        float4 hv = *(const float4*)(h   + (n0 + r) * 128 + dq);
        float4 av = *(const float4*)(agg + (n0 + r) * 128 + dq);
        v4h ph, pa;
        ph[0] = (_Float16)hv.x; ph[1] = (_Float16)hv.y; ph[2] = (_Float16)hv.z; ph[3] = (_Float16)hv.w;
        pa[0] = (_Float16)av.x; pa[1] = (_Float16)av.y; pa[2] = (_Float16)av.z; pa[3] = (_Float16)av.w;
        *(v4h*)(A + r * 256 + dq)       = ph;
        *(v4h*)(A + r * 256 + 128 + dq) = pa;
    }
    __syncthreads();

    const int r0 = (lane >> 4) << 3;
    const int nc = lane & 15;

    // ---- neh = silu([h|agg] @ Wn + bn) ----
    for (int nt = 0; nt < 8; nt += 2) {
        v8f acc0 = {}, acc1 = {};
#pragma unroll
        for (int kt = 0; kt < 8; ++kt) {
            v16h a  = load_frag_a(A, 256, kt * 32, lane);
            v16h b0 = load_frag_b_sw(Wnh, 8, kt, nt, lane);
            v16h b1 = load_frag_b_sw(Wnh, 8, kt, nt + 1, lane);
            acc0 = __builtin_amdgcn_wmma_f32_16x16x32_f16(false, a, false, b0,
                                                          (short)0, acc0, false, false);
            acc1 = __builtin_amdgcn_wmma_f32_16x16x32_f16(false, a, false, b1,
                                                          (short)0, acc1, false, false);
        }
        int c0 = nt * 16 + nc, c1 = c0 + 16;
        float bb0 = bn[c0], bb1 = bn[c1];
#pragma unroll
        for (int e = 0; e < 8; ++e) {
            float v0 = silu_f(acc0[e] + bb0);
            float v1 = silu_f(acc1[e] + bb1);
            C[(r0 + e) * 128 + c0] = v0;
            C[(r0 + e) * 128 + c1] = v1;
            neh_out[(n0 + r0 + e) * 128 + c0] = v0;
            neh_out[(n0 + r0 + e) * 128 + c1] = v1;
        }
    }
    __syncthreads();

    // ---- attention gate; ne -> f16 into reused A [16][128] ----
    if (lane < 16) {
        const float4* cr4 = (const float4*)(C + lane * 128);
        const float4* wa4 = (const float4*)Wa;
        float dot = 0.0f;
        for (int q = 0; q < 32; ++q) {
            float4 t = cr4[q], w = wa4[q];
            dot += t.x * w.x + t.y * w.y + t.z * w.z + t.w * w.w;
        }
        float attn = sigmoid_f(dot + ba[0]);
        _Float16* ar = A + lane * 128;
        for (int q = 0; q < 32; ++q) {
            float4 t = cr4[q];
            v4h o;
            o[0] = (_Float16)(attn * t.x); o[1] = (_Float16)(attn * t.y);
            o[2] = (_Float16)(attn * t.z); o[3] = (_Float16)(attn * t.w);
            *(v4h*)(ar + q * 4) = o;
        }
    }
    __syncthreads();

    // ---- p = silu(ne @ Wp1 + bp1) ----
    for (int nt = 0; nt < 8; nt += 2) {
        v8f acc0 = {}, acc1 = {};
#pragma unroll
        for (int kt = 0; kt < 4; ++kt) {
            v16h a  = load_frag_a(A, 128, kt * 32, lane);
            v16h b0 = load_frag_b_sw(Wp1h, 8, kt, nt, lane);
            v16h b1 = load_frag_b_sw(Wp1h, 8, kt, nt + 1, lane);
            acc0 = __builtin_amdgcn_wmma_f32_16x16x32_f16(false, a, false, b0,
                                                          (short)0, acc0, false, false);
            acc1 = __builtin_amdgcn_wmma_f32_16x16x32_f16(false, a, false, b1,
                                                          (short)0, acc1, false, false);
        }
        int c0 = nt * 16 + nc, c1 = c0 + 16;
        float bb0 = bp1[c0], bb1 = bp1[c1];
#pragma unroll
        for (int e = 0; e < 8; ++e) {
            C[(r0 + e) * 128 + c0] = silu_f(acc0[e] + bb0);
            C[(r0 + e) * 128 + c1] = silu_f(acc1[e] + bb1);
        }
    }
    __syncthreads();

    // ---- LN(128) then dot with Wp2 -> node_energy ----
    if (lane < 16) {
        const float4* cr4 = (const float4*)(C + lane * 128);
        float mu = 0.0f;
        for (int q = 0; q < 32; ++q) {
            float4 t = cr4[q]; mu += t.x + t.y + t.z + t.w;
        }
        mu *= (1.0f / 128.0f);
        float var = 0.0f;
        for (int q = 0; q < 32; ++q) {
            float4 t = cr4[q];
            float a0 = t.x - mu, a1 = t.y - mu, a2 = t.z - mu, a3 = t.w - mu;
            var += a0 * a0 + a1 * a1 + a2 * a2 + a3 * a3;
        }
        var *= (1.0f / 128.0f);
        float inv = rsqrtf(var + LN_EPS);
        const float4* w4 = (const float4*)lnpw;
        const float4* b4 = (const float4*)lnpb;
        const float4* p4 = (const float4*)Wp2;
        float en = 0.0f;
        for (int q = 0; q < 32; ++q) {
            float4 t = cr4[q], w = w4[q], b = b4[q], p = p4[q];
            en += ((t.x - mu) * inv * w.x + b.x) * p.x;
            en += ((t.y - mu) * inv * w.y + b.y) * p.y;
            en += ((t.z - mu) * inv * w.z + b.z) * p.z;
            en += ((t.w - mu) * inv * w.w + b.w) * p.w;
        }
        node_energy[n0 + lane] = en;
    }
}

// ---------------------------------------------------------------------------
// pool: total_energy[b] = silu(node_energy[b,:24] @ Wpool)
// ---------------------------------------------------------------------------
__global__ void pool_kernel(const float* __restrict__ node_energy,
                            const float* __restrict__ Wpool, float* __restrict__ out) {
    int b = blockIdx.x * blockDim.x + threadIdx.x;
    if (b < B_CNT) {
        float s = 0.0f;
#pragma unroll
        for (int n = 0; n < NN_CNT; ++n) s += node_energy[b * NN_CNT + n] * Wpool[n];
        out[b] = silu_f(s);
    }
}

// ---------------------------------------------------------------------------
// launch
// ---------------------------------------------------------------------------
extern "C" void kernel_launch(void* const* d_in, const int* in_sizes, int n_in,
                              void* d_out, int out_size, void* d_ws, size_t ws_size,
                              hipStream_t stream) {
    const float* x    = (const float*)d_in[0];
    const float* h    = (const float*)d_in[1];
    const int*   ei   = (const int*)d_in[2];
    const float* bond = (const float*)d_in[3];
    const float* dis  = (const float*)d_in[4];
    const float* W1   = (const float*)d_in[5];
    const float* b1   = (const float*)d_in[6];
    const float* ln1w = (const float*)d_in[7];
    const float* ln1b = (const float*)d_in[8];
    const float* W2   = (const float*)d_in[9];
    const float* b2   = (const float*)d_in[10];
    const float* ln2w = (const float*)d_in[11];
    const float* ln2b = (const float*)d_in[12];
    const float* Wn   = (const float*)d_in[13];
    const float* bn   = (const float*)d_in[14];
    const float* Wa   = (const float*)d_in[15];
    const float* ba   = (const float*)d_in[16];
    const float* Wp1  = (const float*)d_in[17];
    const float* bp1  = (const float*)d_in[18];
    const float* lnpw = (const float*)d_in[19];
    const float* lnpb = (const float*)d_in[20];
    const float* Wp2  = (const float*)d_in[21];
    const float* Wpool= (const float*)d_in[22];

    float* out        = (float*)d_out;
    float* total_e    = out;                 // [1024]
    float* neh_out    = out + B_CNT;         // [N,128]

    // workspace layout (32B-aligned chunks)
    char* ws = (char*)d_ws;
    _Float16* W1h  = (_Float16*)(ws);                         // 320*256*2 = 163840
    _Float16* W2h  = (_Float16*)(ws + 163840);                // 256*128*2 = 65536
    _Float16* Wnh  = (_Float16*)(ws + 163840 + 65536);        // 65536
    _Float16* Wp1h = (_Float16*)(ws + 163840 + 2 * 65536);    // 128*128*2 = 32768
    float*    agg  = (float*)   (ws + 163840 + 2 * 65536 + 32768);      // N*128*4
    float*    nrg  = (float*)   (ws + 163840 + 2 * 65536 + 32768 + (size_t)N_CNT * 128 * 4);

    // fragment-swizzled f16 weights (W1 padded 313 -> 320 K-rows)
    cvt_swz_kernel<<<(K1_PAD * 256 + 255) / 256, 256, 0, stream>>>(W1, W1h, 313, 256, K1_PAD);
    cvt_swz_kernel<<<(256 * 128 + 255) / 256, 256, 0, stream>>>(W2, W2h, 256, 128, 256);
    cvt_swz_kernel<<<(256 * 128 + 255) / 256, 256, 0, stream>>>(Wn, Wnh, 256, 128, 256);
    cvt_swz_kernel<<<(128 * 128 + 255) / 256, 256, 0, stream>>>(Wp1, Wp1h, 128, 128, 128);

    // zero the aggregation buffer
    zero_kernel<<<(N_CNT * 128 + 255) / 256, 256, 0, stream>>>(agg, N_CNT * 128);

    // edge messages: E/16 tiles, 2 tiles (waves) per block
    edge_msg_kernel<<<E_CNT / 32, 64, 0, stream>>>(x, h, ei, bond, dis,
                                                   W1h, b1, ln1w, ln1b,
                                                   W2h, b2, ln2w, ln2b, agg);

    // node model: N/16 tiles, 2 per block
    node_kernel<<<N_CNT / 32, 64, 0, stream>>>(h, agg, Wnh, bn, Wa, ba,
                                               Wp1h, bp1, lnpw, lnpb, Wp2,
                                               neh_out, nrg);

    // pooling
    pool_kernel<<<(B_CNT + 255) / 256, 256, 0, stream>>>(nrg, Wpool, total_e);
}
